// AutoregressiveTransformer_57629871177837
// MI455X (gfx1250) — matrix operand order repaired and checked by reading, not compiled, not run on hardware
//
#include <hip/hip_runtime.h>
#include <math.h>

typedef __bf16 bf16;
typedef __attribute__((ext_vector_type(16))) __bf16 v16bf;
typedef __attribute__((ext_vector_type(8)))  float  v8f;

#define BB   4
#define SSQ  1024
#define DMM  512
#define NHH  8
#define DHH  512
#define DD   4096
#define LLY  6
#define DFFF 2048
#define KKO  512
#define MT   4096           // B*S rows
#define EPSL 1e-5f

#if __has_builtin(__builtin_amdgcn_global_load_async_to_lds_b128) && \
    __has_builtin(__builtin_amdgcn_s_wait_asynccnt)
#define HAS_ASYNC_LDS 1
#else
#define HAS_ASYNC_LDS 0
#endif

// parameter type of the async builtins: pointer to 16B int vector, AS1 / AS3
typedef int async_v4i __attribute__((vector_size(16)));
#define ASYNC_SRC(p) ((__attribute__((address_space(1))) async_v4i*)(void*)(p))
#define ASYNC_DST(p) ((__attribute__((address_space(3))) async_v4i*)(void*)(p))

union FragBf { v16bf v; uint4 q[2]; };
union Pack8  { uint4 q; bf16 h[8]; };

// ---------------------------------------------------------------------------
// Weight convert + transpose: W [Kd, N] f32 (row-major) -> WT [N, Kd] bf16
// ---------------------------------------------------------------------------
__global__ __launch_bounds__(256) void convT_kernel(const float* __restrict__ W,
                                                    bf16* __restrict__ WT,
                                                    int Kd, int N) {
  int idx = blockIdx.x * 256 + threadIdx.x;
  int total = Kd * N;
  if (idx >= total) return;
  int k = idx % Kd;
  int n = idx / Kd;
  WT[idx] = (bf16)W[(size_t)k * N + n];
}

// ---------------------------------------------------------------------------
// Generic WMMA GEMM: C = A[M,Kd] x Bt[N,Kd]^T (+bias, epilogue by mode)
// 256 threads = 8 waves. Block tile 64(M) x 128(N), BK=32, double-buffered LDS
// with async global->LDS staging (CDNA5) and one barrier per K-step.
// modes: 0 f32+bias | 1 bf16+bias | 2 bf16+bias per-batch-transposed (V^T)
//        3 relu bf16+bias | 4 scores f32 scale+causal (z=b*NH+h) | 5 bf16 (PV)
// ---------------------------------------------------------------------------
__global__ __launch_bounds__(256) void gemm_wmma_kernel(
    const bf16* __restrict__ A, const bf16* __restrict__ Bt,
    const float* __restrict__ bias, float* __restrict__ Cf, bf16* __restrict__ Cb,
    int M, int N, int Kd, int lda, int ldb, int ldc, float scale, int mode)
{
  __shared__ bf16 sA[2][64][32];
  __shared__ bf16 sB[2][128][32];

  const int tid = threadIdx.x;
  const int m0  = blockIdx.y * 64;
  const int n0  = blockIdx.x * 128;
  const int z   = blockIdx.z;

  size_t aOff = 0, bOff = 0, cOff = 0;
  if (mode == 4) {            // scores: A=Q[b,h], Bt=K[b,h], C=S[z]
    int b = z >> 3, h = z & 7;
    aOff = (size_t)b * SSQ * DD + (size_t)h * DHH;
    bOff = aOff;
    cOff = (size_t)z * SSQ * SSQ;
  } else if (mode == 5) {     // PV: A=P[z], Bt=V^T[b,h], C=O[b,h]
    int b = z >> 3, h = z & 7;
    aOff = (size_t)z * SSQ * SSQ;
    bOff = ((size_t)b * DD + (size_t)h * DHH) * SSQ;
    cOff = (size_t)b * SSQ * DD + (size_t)h * DHH;
  }
  const bf16* Ab = A + aOff;
  const bf16* Bb = Bt + bOff;

  const int wv   = tid >> 5, lane = tid & 31;
  const int tm   = (wv & 3) * 16;      // wave row offset within block tile
  const int tn   = (wv >> 2) * 64;     // wave col offset within block tile
  const int nlo  = lane & 15;
  const int hi   = lane >> 4;

  // staging coordinates
  const int arow = tid >> 2;           // 64 rows, one 16B chunk each
  const int acol = (tid & 3) * 8;
  const int brow = tid >> 1;           // 128 rows, two adjacent 16B chunks
  const int bcol = (tid & 1) * 16;

  v8f acc[4];
  #pragma unroll
  for (int i = 0; i < 4; ++i) acc[i] = (v8f){0.f,0.f,0.f,0.f,0.f,0.f,0.f,0.f};

#if HAS_ASYNC_LDS
  auto stage_async = [&](int k0, int buf) {
    __builtin_amdgcn_global_load_async_to_lds_b128(
        ASYNC_SRC(Ab + (size_t)(m0 + arow) * lda + k0 + acol),
        ASYNC_DST(&sA[buf][arow][acol]), 0, 0);
    __builtin_amdgcn_global_load_async_to_lds_b128(
        ASYNC_SRC(Bb + (size_t)(n0 + brow) * ldb + k0 + bcol),
        ASYNC_DST(&sB[buf][brow][bcol]), 0, 0);
    __builtin_amdgcn_global_load_async_to_lds_b128(
        ASYNC_SRC(Bb + (size_t)(n0 + brow) * ldb + k0 + bcol + 8),
        ASYNC_DST(&sB[buf][brow][bcol + 8]), 0, 0);
  };
  // prologue: tile 0
  stage_async(0, 0);
  __builtin_amdgcn_s_wait_asynccnt(0);
  __syncthreads();
#else
  {
    uint4 ra  = *(const uint4*)(Ab + (size_t)(m0 + arow) * lda + acol);
    uint4 rb0 = *(const uint4*)(Bb + (size_t)(n0 + brow) * ldb + bcol);
    uint4 rb1 = *(const uint4*)(Bb + (size_t)(n0 + brow) * ldb + bcol + 8);
    *(uint4*)&sA[0][arow][acol]     = ra;
    *(uint4*)&sB[0][brow][bcol]     = rb0;
    *(uint4*)&sB[0][brow][bcol + 8] = rb1;
  }
  __syncthreads();
#endif

  int buf = 0;
  for (int k0 = 0; k0 < Kd; k0 += 32) {
    const int  nxt  = buf ^ 1;
    const bool more = (k0 + 32 < Kd);

#if HAS_ASYNC_LDS
    if (more) stage_async(k0 + 32, nxt);
#else
    uint4 ra, rb0, rb1;
    if (more) {
      ra  = *(const uint4*)(Ab + (size_t)(m0 + arow) * lda + (k0 + 32) + acol);
      rb0 = *(const uint4*)(Bb + (size_t)(n0 + brow) * ldb + (k0 + 32) + bcol);
      rb1 = *(const uint4*)(Bb + (size_t)(n0 + brow) * ldb + (k0 + 32) + bcol + 8);
    }
#endif

    // compute from buf
    // A fragment (16x32): lanes 0-15 rows m, K 0-7 & 16-23; lanes 16-31 K 8-15 & 24-31
    FragBf fa;
    {
      int mrow = tm + nlo;
      int koff = hi * 8;
      fa.q[0] = *(uint4*)&sA[buf][mrow][koff];
      fa.q[1] = *(uint4*)&sA[buf][mrow][koff + 16];
    }
    #pragma unroll
    for (int nt = 0; nt < 4; ++nt) {
      // B fragment (32x16): lanes 0-15 K 0-7 & 16-23; lanes 16-31 K 8-15 & 24-31
      FragBf fb;
      int col = tn + nt * 16 + nlo;
      int kb  = hi * 8;
      fb.q[0] = *(uint4*)&sB[buf][col][kb];
      fb.q[1] = *(uint4*)&sB[buf][col][kb + 16];
      acc[nt] = __builtin_amdgcn_wmma_f32_16x16x32_bf16(
          false, fa.v, false, fb.v, (short)0, acc[nt], false, false);
    }

    if (more) {
#if HAS_ASYNC_LDS
      __builtin_amdgcn_s_wait_asynccnt(0);
#else
      *(uint4*)&sA[nxt][arow][acol]     = ra;
      *(uint4*)&sB[nxt][brow][bcol]     = rb0;
      *(uint4*)&sB[nxt][brow][bcol + 8] = rb1;
#endif
      __syncthreads();
    }
    buf = nxt;
  }

  // epilogue
  const int gm0 = m0 + tm + hi * 8;   // 8 consecutive M values per lane (vgpr idx)
  #pragma unroll
  for (int nt = 0; nt < 4; ++nt) {
    int gn = n0 + tn + nt * 16 + nlo;
    float bv = (mode <= 3) ? bias[gn] : 0.0f;
    if (mode == 0) {
      #pragma unroll
      for (int i = 0; i < 8; ++i)
        Cf[cOff + (size_t)(gm0 + i) * ldc + gn] = acc[nt][i] + bv;
    } else if (mode == 1) {
      #pragma unroll
      for (int i = 0; i < 8; ++i)
        Cb[cOff + (size_t)(gm0 + i) * ldc + gn] = (bf16)(acc[nt][i] + bv);
    } else if (mode == 2) {
      // transposed per-batch store: VT[((b*DD)+gn)*S + t], 8 contiguous t
      Pack8 p;
      #pragma unroll
      for (int i = 0; i < 8; ++i) p.h[i] = (bf16)(acc[nt][i] + bv);
      int b = gm0 >> 10, t = gm0 & 1023;
      *(uint4*)&Cb[((size_t)b * DD + gn) * SSQ + t] = p.q;
    } else if (mode == 3) {
      #pragma unroll
      for (int i = 0; i < 8; ++i) {
        float v = acc[nt][i] + bv;
        Cb[cOff + (size_t)(gm0 + i) * ldc + gn] = (bf16)(v > 0.f ? v : 0.f);
      }
    } else if (mode == 4) {
      #pragma unroll
      for (int i = 0; i < 8; ++i) {
        int gm = gm0 + i;
        float v = acc[nt][i] * scale;
        if (gn > gm) v = -1e30f;               // causal mask
        Cf[cOff + (size_t)gm * ldc + gn] = v;
      }
    } else {
      #pragma unroll
      for (int i = 0; i < 8; ++i)
        Cb[cOff + (size_t)(gm0 + i) * ldc + gn] = (bf16)acc[nt][i];
    }
  }
}

// ---------------------------------------------------------------------------
// Row softmax over S=1024 fp32 scores -> bf16 probabilities
// ---------------------------------------------------------------------------
__global__ __launch_bounds__(256) void softmax_kernel(const float* __restrict__ S_,
                                                      bf16* __restrict__ P) {
  __shared__ float red[256];
  const size_t row = blockIdx.x;
  const float* sp = S_ + row * SSQ;
  bf16* pp = P + row * SSQ;
  const int tid = threadIdx.x;

  float v[4];
  float mx = -3.4e38f;
  #pragma unroll
  for (int i = 0; i < 4; ++i) { v[i] = sp[tid + i * 256]; mx = fmaxf(mx, v[i]); }
  red[tid] = mx; __syncthreads();
  for (int s = 128; s > 0; s >>= 1) {
    if (tid < s) red[tid] = fmaxf(red[tid], red[tid + s]);
    __syncthreads();
  }
  mx = red[0]; __syncthreads();

  float e[4]; float sum = 0.f;
  #pragma unroll
  for (int i = 0; i < 4; ++i) { e[i] = expf(v[i] - mx); sum += e[i]; }
  red[tid] = sum; __syncthreads();
  for (int s = 128; s > 0; s >>= 1) {
    if (tid < s) red[tid] += red[tid + s];
    __syncthreads();
  }
  float inv = 1.0f / red[0];
  #pragma unroll
  for (int i = 0; i < 4; ++i) pp[tid + i * 256] = (bf16)(e[i] * inv);
}

// ---------------------------------------------------------------------------
// Embedding + input LayerNorm: h = LN(tok[x]+pos)*g + b, emits f32 + bf16
// ---------------------------------------------------------------------------
__global__ __launch_bounds__(128) void embed_ln_kernel(
    const int* __restrict__ x, const float* __restrict__ tok,
    const float* __restrict__ pos, const float* __restrict__ g,
    const float* __restrict__ bta, float* __restrict__ h, bf16* __restrict__ hb) {
  __shared__ float red[128];
  const int r = blockIdx.x;
  const int s = r & 1023;
  const int tid = threadIdx.x;
  const int tokid = x[r];

  float v[4]; float sum = 0.f;
  #pragma unroll
  for (int i = 0; i < 4; ++i) {
    int c = tid + i * 128;
    float t = tok[(size_t)tokid * DMM + c] + pos[(size_t)s * DMM + c];
    v[i] = t; sum += t;
  }
  red[tid] = sum; __syncthreads();
  for (int st = 64; st > 0; st >>= 1) { if (tid < st) red[tid] += red[tid + st]; __syncthreads(); }
  float mean = red[0] * (1.0f / DMM); __syncthreads();

  float sq = 0.f;
  #pragma unroll
  for (int i = 0; i < 4; ++i) { float d = v[i] - mean; sq += d * d; }
  red[tid] = sq; __syncthreads();
  for (int st = 64; st > 0; st >>= 1) { if (tid < st) red[tid] += red[tid + st]; __syncthreads(); }
  float inv = rsqrtf(red[0] * (1.0f / DMM) + EPSL);

  #pragma unroll
  for (int i = 0; i < 4; ++i) {
    int c = tid + i * 128;
    float y = (v[i] - mean) * inv * g[c] + bta[c];
    h[(size_t)r * DMM + c]  = y;
    hb[(size_t)r * DMM + c] = (bf16)y;
  }
}

// ---------------------------------------------------------------------------
// Residual + LayerNorm: h = LN(a + h)*g + b (in place on h), emits bf16 copy
// ---------------------------------------------------------------------------
__global__ __launch_bounds__(128) void resid_ln_kernel(
    const float* __restrict__ a, float* __restrict__ h, bf16* __restrict__ hb,
    const float* __restrict__ g, const float* __restrict__ bta) {
  __shared__ float red[128];
  const int r = blockIdx.x;
  const int tid = threadIdx.x;

  float v[4]; float sum = 0.f;
  #pragma unroll
  for (int i = 0; i < 4; ++i) {
    int c = tid + i * 128;
    float t = a[(size_t)r * DMM + c] + h[(size_t)r * DMM + c];
    v[i] = t; sum += t;
  }
  red[tid] = sum; __syncthreads();
  for (int st = 64; st > 0; st >>= 1) { if (tid < st) red[tid] += red[tid + st]; __syncthreads(); }
  float mean = red[0] * (1.0f / DMM); __syncthreads();

  float sq = 0.f;
  #pragma unroll
  for (int i = 0; i < 4; ++i) { float d = v[i] - mean; sq += d * d; }
  red[tid] = sq; __syncthreads();
  for (int st = 64; st > 0; st >>= 1) { if (tid < st) red[tid] += red[tid + st]; __syncthreads(); }
  float inv = rsqrtf(red[0] * (1.0f / DMM) + EPSL);

  #pragma unroll
  for (int i = 0; i < 4; ++i) {
    int c = tid + i * 128;
    float y = (v[i] - mean) * inv * g[c] + bta[c];
    h[(size_t)r * DMM + c]  = y;
    hb[(size_t)r * DMM + c] = (bf16)y;
  }
}

// ---------------------------------------------------------------------------
// Host orchestration
// ---------------------------------------------------------------------------
extern "C" void kernel_launch(void* const* d_in, const int* in_sizes, int n_in,
                              void* d_out, int out_size, void* d_ws, size_t ws_size,
                              hipStream_t stream) {
  (void)in_sizes; (void)n_in; (void)out_size; (void)ws_size;

  const int*   x       = (const int*)  d_in[0];
  const float* tok_emb = (const float*)d_in[1];
  const float* pos_emb = (const float*)d_in[2];
  const float* ln_in_g = (const float*)d_in[3];
  const float* ln_in_b = (const float*)d_in[4];
  const float* Wq = (const float*)d_in[5];  const float* bq = (const float*)d_in[6];
  const float* Wk = (const float*)d_in[7];  const float* bk = (const float*)d_in[8];
  const float* Wv = (const float*)d_in[9];  const float* bv = (const float*)d_in[10];
  const float* Wo = (const float*)d_in[11]; const float* bo = (const float*)d_in[12];
  const float* ln1g = (const float*)d_in[13]; const float* ln1b = (const float*)d_in[14];
  const float* W1 = (const float*)d_in[15]; const float* b1 = (const float*)d_in[16];
  const float* W2 = (const float*)d_in[17]; const float* b2 = (const float*)d_in[18];
  const float* ln2g = (const float*)d_in[19]; const float* ln2b = (const float*)d_in[20];
  const float* Wout = (const float*)d_in[21]; const float* bout = (const float*)d_in[22];

  // workspace bump allocator
  char* base = (char*)d_ws;
  size_t off = 0;
  auto alloc = [&](size_t bytes) -> void* {
    void* p = base + off;
    off = (off + bytes + 255) & ~(size_t)255;
    return p;
  };

  float* h    = (float*)alloc((size_t)MT * DMM * 4);
  bf16*  hb   = (bf16*) alloc((size_t)MT * DMM * 2);
  bf16*  WqT  = (bf16*) alloc((size_t)LLY * DD * DMM * 2);
  bf16*  WkT  = (bf16*) alloc((size_t)LLY * DD * DMM * 2);
  bf16*  WvT  = (bf16*) alloc((size_t)LLY * DD * DMM * 2);
  bf16*  WoT  = (bf16*) alloc((size_t)LLY * DMM * DD * 2);
  bf16*  W1T  = (bf16*) alloc((size_t)LLY * DFFF * DMM * 2);
  bf16*  W2T  = (bf16*) alloc((size_t)LLY * DMM * DFFF * 2);
  bf16*  WoutT= (bf16*) alloc((size_t)KKO * DMM * 2);
  bf16*  Qb   = (bf16*) alloc((size_t)MT * DD * 2);
  bf16*  Kb   = (bf16*) alloc((size_t)MT * DD * 2);
  bf16*  VT   = (bf16*) alloc((size_t)BB * DD * SSQ * 2);
  float* Sc   = (float*)alloc((size_t)BB * NHH * SSQ * SSQ * 4);
  bf16*  Pb   = (bf16*) alloc((size_t)BB * NHH * SSQ * SSQ * 2);
  bf16*  Ob   = (bf16*) alloc((size_t)MT * DD * 2);
  float* proj = (float*)alloc((size_t)MT * DMM * 4);
  bf16*  ff1  = (bf16*) alloc((size_t)MT * DFFF * 2);

  auto convT = [&](const float* W, bf16* WT, int Kd, int N) {
    int total = Kd * N;
    convT_kernel<<<(total + 255) / 256, 256, 0, stream>>>(W, WT, Kd, N);
  };
  auto gemm = [&](const bf16* A, const bf16* Bt, const float* bias,
                  float* Cf, bf16* Cb, int M, int N, int Kd,
                  int lda, int ldb, int ldc, float scale, int mode, int Z) {
    dim3 grid(N / 128, M / 64, Z);
    gemm_wmma_kernel<<<grid, 256, 0, stream>>>(A, Bt, bias, Cf, Cb,
                                               M, N, Kd, lda, ldb, ldc, scale, mode);
  };

  // weight conversion / transpose (bf16, K-contiguous per output column)
  for (int l = 0; l < LLY; ++l) {
    convT(Wq + (size_t)l * DMM * DD,   WqT + (size_t)l * DD * DMM,   DMM,  DD);
    convT(Wk + (size_t)l * DMM * DD,   WkT + (size_t)l * DD * DMM,   DMM,  DD);
    convT(Wv + (size_t)l * DMM * DD,   WvT + (size_t)l * DD * DMM,   DMM,  DD);
    convT(Wo + (size_t)l * DD * DMM,   WoT + (size_t)l * DMM * DD,   DD,   DMM);
    convT(W1 + (size_t)l * DMM * DFFF, W1T + (size_t)l * DFFF * DMM, DMM,  DFFF);
    convT(W2 + (size_t)l * DFFF * DMM, W2T + (size_t)l * DMM * DFFF, DFFF, DMM);
  }
  convT(Wout, WoutT, DMM, KKO);

  // embedding + input LN
  embed_ln_kernel<<<MT, 128, 0, stream>>>(x, tok_emb, pos_emb, ln_in_g, ln_in_b, h, hb);

  const float scl = 0.044194173824159216f;  // 1/sqrt(512)

  for (int l = 0; l < LLY; ++l) {
    // Q, K projections (bf16 row-major), V projection stored transposed per batch
    gemm(hb, WqT + (size_t)l * DD * DMM, bq + (size_t)l * DD, nullptr, Qb,
         MT, DD, DMM, DMM, DMM, DD, 1.f, 1, 1);
    gemm(hb, WkT + (size_t)l * DD * DMM, bk + (size_t)l * DD, nullptr, Kb,
         MT, DD, DMM, DMM, DMM, DD, 1.f, 1, 1);
    gemm(hb, WvT + (size_t)l * DD * DMM, bv + (size_t)l * DD, nullptr, VT,
         MT, DD, DMM, DMM, DMM, DD, 1.f, 2, 1);
    // scores = scale * Q K^T + causal mask   (per b,h)
    gemm(Qb, Kb, nullptr, Sc, nullptr,
         SSQ, SSQ, DHH, DD, DD, SSQ, scl, 4, BB * NHH);
    // softmax rows -> bf16 probs
    softmax_kernel<<<BB * NHH * SSQ, 256, 0, stream>>>(Sc, Pb);
    // O = P V      (per b,h; V^T gives contiguous B operand)
    gemm(Pb, VT, nullptr, nullptr, Ob,
         SSQ, DHH, SSQ, SSQ, SSQ, DD, 1.f, 5, BB * NHH);
    // attn output projection (f32) + residual + LN1
    gemm(Ob, WoT + (size_t)l * DMM * DD, bo + (size_t)l * DMM, proj, nullptr,
         MT, DMM, DD, DD, DD, DMM, 1.f, 0, 1);
    resid_ln_kernel<<<MT, 128, 0, stream>>>(proj, h, hb,
                                            ln1g + (size_t)l * DMM, ln1b + (size_t)l * DMM);
    // FFN
    gemm(hb, W1T + (size_t)l * DFFF * DMM, b1 + (size_t)l * DFFF, nullptr, ff1,
         MT, DFFF, DMM, DMM, DMM, DFFF, 1.f, 3, 1);
    gemm(ff1, W2T + (size_t)l * DMM * DFFF, b2 + (size_t)l * DMM, proj, nullptr,
         MT, DMM, DFFF, DFFF, DFFF, DMM, 1.f, 0, 1);
    resid_ln_kernel<<<MT, 128, 0, stream>>>(proj, h, hb,
                                            ln2g + (size_t)l * DMM, ln2b + (size_t)l * DMM);
  }

  // final head: logits = h @ W_out + b_out  (f32, straight to d_out)
  gemm(hb, WoutT, bout, (float*)d_out, nullptr,
       MT, KKO, DMM, DMM, DMM, KKO, 1.f, 0, 1);
}
